// AttentionFusion_70385924046872
// MI455X (gfx1250) — matrix operand order repaired and checked by reading, not compiled, not run on hardware
//
#include <hip/hip_runtime.h>
#include <hip/hip_bf16.h>
#include <math.h>

// ---------------------------------------------------------------------------
// AttentionFusion for MI455X (gfx1250, wave32, WMMA bf16 + Tensor Data Mover).
//
// Algebra: scores = X A X^T (+bias terms), A = Wq^T Wk;  out = xmix Wc^T + bc,
// Wc = Wo Wv, bc = Wo bv + bo, xmix = conf-softmax-mean-weighted X rows.
// 90 GFLOP total (vs 224 naive), compute-bound -> v_wmma_f32_16x16x32_bf16,
// LDS tiles staged by TENSOR_LOAD_TO_LDS (TDM) with double buffering.
// ---------------------------------------------------------------------------

#define Hdim 1024
#define Bdim 8192
#define Mdim 4

typedef __attribute__((ext_vector_type(16))) __bf16 v16bf;
typedef __attribute__((ext_vector_type(8)))  float  v8f;
typedef __attribute__((ext_vector_type(4)))  unsigned int v4u;
typedef __attribute__((ext_vector_type(8)))  int v8i;
typedef __attribute__((ext_vector_type(4)))  int v4i;

#if __has_builtin(__builtin_amdgcn_tensor_load_to_lds) && \
    __has_builtin(__builtin_amdgcn_s_wait_tensorcnt)
#define USE_TDM 1
#else
#define USE_TDM 0
#endif

union Frag16 {
  v16bf          bf;
  unsigned short u[16];
  uint4          q[2];
};

__device__ __forceinline__ unsigned short f32_to_bf16(float f) {
  union { float f; unsigned u; } v; v.f = f;
  unsigned r = (v.u + 0x7FFFu + ((v.u >> 16) & 1u)) >> 16;   // RNE
  return (unsigned short)r;
}
__device__ __forceinline__ float bf16_to_f32(unsigned short h) {
  return __uint_as_float(((unsigned)h) << 16);
}
__device__ __forceinline__ v8f vzero8() {
  v8f z = {0.f,0.f,0.f,0.f,0.f,0.f,0.f,0.f};
  return z;
}

typedef __attribute__((address_space(3))) const void* lds_cptr_t;
__device__ __forceinline__ unsigned lds_addr_of(const void* p) {
  return (unsigned)(unsigned long long)(lds_cptr_t)p;   // raw LDS byte offset
}

// A-fragment (16x32, 16-bit): lane<16 holds M=lane, K={0..7, 16..23};
// lane>=16 holds M=lane-16, K={8..15, 24..31}.
__device__ __forceinline__ v16bf load_a_frag(const unsigned short* src, int ld,
                                             int mbase, int lane) {
  const int row = mbase + (lane & 15);
  const int kb  = (lane >> 4) << 3;
  const unsigned short* p = src + row * ld + kb;
  Frag16 f;
  f.q[0] = *(const uint4*)(p);        // K = kb .. kb+7
  f.q[1] = *(const uint4*)(p + 16);   // K = kb+16 .. kb+23
  return f.bf;
}

// B-fragment (32x16): lane<16 holds N=lane, K={0..7, 16..23}; lane>=16 N=lane-16, K={8..15, 24..31}.
__device__ __forceinline__ v16bf load_b_frag(const unsigned short* src, int ld,
                                             int nbase, int lane) {
  const int row = nbase + (lane & 15);
  const int kb  = (lane >> 4) << 3;
  const unsigned short* p = src + row * ld + kb;
  Frag16 f;
  f.q[0] = *(const uint4*)(p);
  f.q[1] = *(const uint4*)(p + 16);
  return f.bf;
}

#if USE_TDM
// DMA a 128-row x 32-col bf16 tile (row-major, row length = rowlen elements)
// into LDS at lds_byte_addr with row stride 40 halves (pad: every 16 DWORDs
// stored, insert 4 DWORDs). D# per CDNA5 ISA.
// This toolchain's builtin takes 6 args: (g0, g1, g2, g3, extra v8i, cpol).
__device__ __forceinline__ void tdm_load_tile_128x32(
    const unsigned short* gbase, unsigned long long elem_off,
    unsigned lds_byte_addr, unsigned tensor_rows, unsigned rowlen) {
  unsigned long long ga = (unsigned long long)gbase + elem_off * 2ull;
  v4u g0;
  g0[0] = 1u;                                             // count=1 (valid)
  g0[1] = lds_byte_addr;                                  // lds_addr
  g0[2] = (unsigned)(ga & 0xFFFFFFFFull);                 // global_addr lo
  g0[3] = (unsigned)((ga >> 32) & 0x01FFFFFFull)          // global_addr[56:32]
          | (2u << 30);                                   // type=2 ("image")
  v8i g1;
  g1[0] = (int)((1u << 16)                                // data_size=1 -> 2 bytes
              | (1u << 20)                                // pad_enable
              | (3u << 22)                                // pad_interval: 16 DWORDs
              | (3u << 25));                              // pad_amount: 4 DWORDs
  g1[1] = (int)((rowlen & 0xFFFFu) << 16);                // tensor_dim0 lo16 @[63:48]
  g1[2] = (int)(((rowlen >> 16) & 0xFFFFu)                // tensor_dim0 hi16 @[79:64]
              | ((tensor_rows & 0xFFFFu) << 16));         // tensor_dim1 lo16 @[95:80]
  g1[3] = (int)(((tensor_rows >> 16) & 0xFFFFu)           // tensor_dim1 hi16 @[111:96]
              | (32u << 16));                             // tile_dim0=32 @[127:112]
  g1[4] = (int)128u;                                      // tile_dim1=128, tile_dim2=0
  g1[5] = (int)rowlen;                                    // tensor_dim0_stride lo32
  g1[6] = 0;                                              // stride hi16 | dim1_stride lo
  g1[7] = 0;
  v4i gz4 = {0, 0, 0, 0};
  v8i gz8 = {0, 0, 0, 0, 0, 0, 0, 0};
  __builtin_amdgcn_tensor_load_to_lds(g0, g1, gz4, gz4, gz8, 0);
}
#endif

// ---------------------------------------------------------------------------
// Tiled GEMM:  C[M,N] = Asrc[M,K] * Bsrc[N,K]^T  (bf16 operands, K-contiguous)
// Block 256 threads = 8 waves (2x4), block tile 128x128, wave tile 64x32.
// Double-buffered LDS; tiles staged by TDM (fallback: vector loads/stores).
// OUT_F32_BIAS: write fp32 + bias[col]; else write bf16.
// ---------------------------------------------------------------------------
template <bool OUT_F32_BIAS>
__global__ __launch_bounds__(256)
void gemm_nt_kernel(const unsigned short* __restrict__ Asrc,
                    const unsigned short* __restrict__ Bsrc,
                    void* __restrict__ Cdst,
                    const float* __restrict__ bias,
                    int M, int N, int K) {
  __shared__ unsigned short Asb[2][128][40];  // 80B row stride: aligned, conflict-free
  __shared__ unsigned short Bsb[2][128][40];

  const int tid  = threadIdx.x;
  const int lane = tid & 31;
  const int wave = tid >> 5;
  const int wr   = wave >> 2;     // 0..1  (64 rows each)
  const int wc   = wave & 3;      // 0..3  (32 cols each)
  const int bm   = blockIdx.x * 128;
  const int bn   = blockIdx.y * 128;
  const int nchunks = K >> 5;

  auto issue_chunk = [&](int ck, int buf) {
#if USE_TDM
    if (wave == 0) {   // one wave issues the DMA; EXEC-independent, TENSORcnt-tracked
      tdm_load_tile_128x32(Asrc, (unsigned long long)bm * K + (unsigned)(ck * 32),
                           lds_addr_of(&Asb[buf][0][0]), (unsigned)M, (unsigned)K);
      tdm_load_tile_128x32(Bsrc, (unsigned long long)bn * K + (unsigned)(ck * 32),
                           lds_addr_of(&Bsb[buf][0][0]), (unsigned)N, (unsigned)K);
    }
#else
#pragma unroll
    for (int i = 0; i < 2; ++i) {
      int lin = tid + 256 * i;            // uint4 (8 halves) index over 512
      int r = lin >> 2, c = (lin & 3) << 3;
      *(uint4*)&Asb[buf][r][c] = *(const uint4*)(Asrc + (size_t)(bm + r) * K + ck * 32 + c);
      *(uint4*)&Bsb[buf][r][c] = *(const uint4*)(Bsrc + (size_t)(bn + r) * K + ck * 32 + c);
    }
    if (ck + 1 < nchunks)
      __builtin_prefetch((const char*)(Bsrc + (size_t)(bn + (tid >> 1)) * K + ck * 32 + 32), 0, 1);
#endif
  };

  v8f acc[4][2];
#pragma unroll
  for (int mt = 0; mt < 4; ++mt)
#pragma unroll
    for (int nt = 0; nt < 2; ++nt) acc[mt][nt] = vzero8();

  issue_chunk(0, 0);

  for (int ck = 0; ck < nchunks; ++ck) {
    const int cur = ck & 1;
    if (ck + 1 < nchunks) {
      issue_chunk(ck + 1, cur ^ 1);       // overlap next DMA with this compute
#if USE_TDM
      if (wave == 0) __builtin_amdgcn_s_wait_tensorcnt(2);  // chunk ck resident
#endif
    } else {
#if USE_TDM
      if (wave == 0) __builtin_amdgcn_s_wait_tensorcnt(0);
#endif
    }
    __syncthreads();                      // publish tile ck to all waves

    v16bf bfr[2];
#pragma unroll
    for (int nt = 0; nt < 2; ++nt)
      bfr[nt] = load_b_frag(&Bsb[cur][0][0], 40, wc * 32 + nt * 16, lane);
#pragma unroll
    for (int mt = 0; mt < 4; ++mt) {
      v16bf afr = load_a_frag(&Asb[cur][0][0], 40, wr * 64 + mt * 16, lane);
#pragma unroll
      for (int nt = 0; nt < 2; ++nt) {
        acc[mt][nt] = __builtin_amdgcn_wmma_f32_16x16x32_bf16(
            false, afr, false, bfr[nt], (short)0, acc[mt][nt], false, false);
      }
    }
    __syncthreads();                      // reads done before buffer reuse
  }

  // epilogue: D layout = VGPR j: (lanes 0-15 -> M=j, N=lane),
  //                      (lanes 16-31 -> M=8+j, N=lane-16)
  const int rofs = ((lane >> 4) << 3);
  const int col0 = lane & 15;
#pragma unroll
  for (int mt = 0; mt < 4; ++mt)
#pragma unroll
    for (int nt = 0; nt < 2; ++nt)
#pragma unroll
      for (int j = 0; j < 8; ++j) {
        int row = bm + wr * 64 + mt * 16 + j + rofs;
        int col = bn + wc * 32 + nt * 16 + col0;
        float v = acc[mt][nt][j];
        if (OUT_F32_BIAS) {
          ((float*)Cdst)[(size_t)row * N + col] = v + bias[col];
        } else {
          ((unsigned short*)Cdst)[(size_t)row * N + col] = f32_to_bf16(v);
        }
      }
}

// ---------------------------------------------------------------------------
// fp32 -> bf16 conversion kernels (optionally transposed).
// ---------------------------------------------------------------------------
__global__ void cvt_transpose_kernel(const float* __restrict__ W,
                                     unsigned short* __restrict__ T) {
  int idx = blockIdx.x * 256 + threadIdx.x;   // over H*H
  int o = idx >> 10, h = idx & (Hdim - 1);
  T[h * Hdim + o] = f32_to_bf16(W[idx]);      // T[h][o] = W[o][h]
}
__global__ void cvt_copy_kernel(const float* __restrict__ W,
                                unsigned short* __restrict__ T) {
  int idx = blockIdx.x * 256 + threadIdx.x;
  T[idx] = f32_to_bf16(W[idx]);
}

// u[h] = Wq^T bk, w[h] = Wk^T bq, bc[j] = Wo bv + bo, c0 = bq.bk  (tiny)
__global__ void vec_pre_kernel(const float* __restrict__ Wq, const float* __restrict__ Wk,
                               const float* __restrict__ Wo,
                               const float* __restrict__ bq, const float* __restrict__ bk,
                               const float* __restrict__ bv, const float* __restrict__ bo,
                               float* __restrict__ u, float* __restrict__ w,
                               float* __restrict__ bc, float* __restrict__ c0) {
  int i = blockIdx.x * 256 + threadIdx.x;     // 0..1023
  float su = 0.f, sw = 0.f;
  for (int o = 0; o < Hdim; ++o) {
    su += Wq[o * Hdim + i] * bk[o];
    sw += Wk[o * Hdim + i] * bq[o];
  }
  u[i] = su; w[i] = sw;
  float sb = 0.f;
  for (int h = 0; h < Hdim; ++h) sb += Wo[i * Hdim + h] * bv[h];
  bc[i] = sb + bo[i];
  if (i == 0) {
    float s = 0.f;
    for (int o = 0; o < Hdim; ++o) s += bq[o] * bk[o];
    *c0 = s;
  }
}

// ---------------------------------------------------------------------------
// Attention: one wave per batch. scores[m,n] = X_m.R_n + du_m + dw_n + c0,
// scale 1/32, * conf_m, softmax over n, wsum_n = mean_m attn, xmix = sum wsum X.
// ---------------------------------------------------------------------------
__device__ __forceinline__ float wave_sum(float v) {
#pragma unroll
  for (int off = 16; off > 0; off >>= 1) v += __shfl_xor(v, off, 32);
  return v;
}

__global__ __launch_bounds__(256)
void attn_kernel(const float* __restrict__ X, const float* __restrict__ conf,
                 const unsigned short* __restrict__ R,
                 const float* __restrict__ u, const float* __restrict__ w,
                 const float* __restrict__ c0p,
                 unsigned short* __restrict__ xmix) {
  const int lane = threadIdx.x & 31;
  const int wave = threadIdx.x >> 5;
  const int b    = blockIdx.x * 8 + wave;
  const float* Xb          = X + (size_t)b * Mdim * Hdim;
  const unsigned short* Rb = R + (size_t)b * Mdim * Hdim;

  float S[4][4] = {{0}}, du[4] = {0}, dw[4] = {0};
  for (int it = 0; it < 32; ++it) {
    int h = lane + 32 * it;
    float x[4], r[4];
#pragma unroll
    for (int m = 0; m < 4; ++m) x[m] = Xb[m * Hdim + h];
#pragma unroll
    for (int n = 0; n < 4; ++n) r[n] = bf16_to_f32(Rb[n * Hdim + h]);
    float uh = u[h], wh = w[h];
#pragma unroll
    for (int m = 0; m < 4; ++m) {
#pragma unroll
      for (int n = 0; n < 4; ++n) S[m][n] += x[m] * r[n];
      du[m] += x[m] * uh;
      dw[m] += x[m] * wh;
    }
  }
#pragma unroll
  for (int m = 0; m < 4; ++m) {
#pragma unroll
    for (int n = 0; n < 4; ++n) S[m][n] = wave_sum(S[m][n]);
    du[m] = wave_sum(du[m]);
    dw[m] = wave_sum(dw[m]);
  }

  const float C0 = *c0p;
  float wsum[4] = {0.f, 0.f, 0.f, 0.f};
#pragma unroll
  for (int m = 0; m < 4; ++m) {
    float cm = conf[(size_t)b * Mdim + m];
    float sc[4];
#pragma unroll
    for (int n = 0; n < 4; ++n)
      sc[n] = (S[m][n] + du[m] + dw[n] + C0) * 0.03125f * cm;  // 1/sqrt(1024)
    float mx = fmaxf(fmaxf(sc[0], sc[1]), fmaxf(sc[2], sc[3]));
    float e[4], se = 0.f;
#pragma unroll
    for (int n = 0; n < 4; ++n) { e[n] = __expf(sc[n] - mx); se += e[n]; }
    float inv = 0.25f / se;                                    // mean over m folded in
#pragma unroll
    for (int n = 0; n < 4; ++n) wsum[n] += e[n] * inv;
  }

  for (int it = 0; it < 32; ++it) {
    int h = lane + 32 * it;
    float acc = 0.f;
#pragma unroll
    for (int n = 0; n < 4; ++n) acc += wsum[n] * Xb[n * Hdim + h];
    xmix[(size_t)b * Hdim + h] = f32_to_bf16(acc);
  }
}

// ---------------------------------------------------------------------------
// Workspace layout (~156.1 MB):
//   [0,2M)  WqT   [2M,4M)  WkT   [4M,6M)  WvT   [6M,8M)  Wo bf16
//   [8M,10M) A=Wq^T Wk  [10M,12M) Wc=Wo Wv  [12M, 12M+16K) u,w,bc,c0
//   [12M+16K, +64M) R bf16[32768,1024]  [76M+16K, +16M) xmix bf16
//   [92M+16K, +64M) Xbf bf16[32768,1024]
// ---------------------------------------------------------------------------
extern "C" void kernel_launch(void* const* d_in, const int* in_sizes, int n_in,
                              void* d_out, int out_size, void* d_ws, size_t ws_size,
                              hipStream_t stream) {
  const float* features = (const float*)d_in[0];   // [8192,4,1024]
  const float* conf     = (const float*)d_in[1];   // [8192,4,1]
  const float* Wq = (const float*)d_in[2];
  const float* bq = (const float*)d_in[3];
  const float* Wk = (const float*)d_in[4];
  const float* bk = (const float*)d_in[5];
  const float* Wv = (const float*)d_in[6];
  const float* bv = (const float*)d_in[7];
  const float* Wo = (const float*)d_in[8];
  const float* bo = (const float*)d_in[9];
  float* out = (float*)d_out;                       // [8192,1024]

  char* ws = (char*)d_ws;
  const size_t MB = 1024ull * 1024ull;
  unsigned short* WqT  = (unsigned short*)(ws + 0 * MB);
  unsigned short* WkT  = (unsigned short*)(ws + 2 * MB);
  unsigned short* WvT  = (unsigned short*)(ws + 4 * MB);
  unsigned short* Wob  = (unsigned short*)(ws + 6 * MB);
  unsigned short* Amat = (unsigned short*)(ws + 8 * MB);
  unsigned short* Wc   = (unsigned short*)(ws + 10 * MB);
  float* u  = (float*)(ws + 12 * MB);
  float* w  = (float*)(ws + 12 * MB + 4096);
  float* bc = (float*)(ws + 12 * MB + 8192);
  float* c0 = (float*)(ws + 12 * MB + 12288);
  unsigned short* Rws  = (unsigned short*)(ws + 12 * MB + 16384);   // 64 MB
  unsigned short* xmix = (unsigned short*)(ws + 76 * MB + 16384);   // 16 MB
  unsigned short* Xbf  = (unsigned short*)(ws + 92 * MB + 16384);   // 64 MB

  // 1) weight + feature conversions to bf16
  cvt_transpose_kernel<<<4096, 256, 0, stream>>>(Wq, WqT);
  cvt_transpose_kernel<<<4096, 256, 0, stream>>>(Wk, WkT);
  cvt_transpose_kernel<<<4096, 256, 0, stream>>>(Wv, WvT);
  cvt_copy_kernel<<<4096, 256, 0, stream>>>(Wo, Wob);
  cvt_copy_kernel<<<(Bdim * Mdim * Hdim) / 256, 256, 0, stream>>>(features, Xbf);

  // 2) bias-fold vectors
  vec_pre_kernel<<<4, 256, 0, stream>>>(Wq, Wk, Wo, bq, bk, bv, bo, u, w, bc, c0);

  // 3) A = Wq^T Wk  and  Wc = Wo Wv   (1024^3, bf16 out)
  gemm_nt_kernel<false><<<dim3(8, 8), 256, 0, stream>>>(WqT, WkT, Amat, nullptr,
                                                        Hdim, Hdim, Hdim);
  gemm_nt_kernel<false><<<dim3(8, 8), 256, 0, stream>>>(Wob, WvT, Wc, nullptr,
                                                        Hdim, Hdim, Hdim);

  // 4) R = X A^T   (32768 x 1024 x 1024)
  gemm_nt_kernel<false><<<dim3(256, 8), 256, 0, stream>>>(Xbf, Amat, Rws, nullptr,
                                                          Bdim * Mdim, Hdim, Hdim);

  // 5) attention: scores / softmax / wsum / xmix  (one wave per batch)
  attn_kernel<<<Bdim / 8, 256, 0, stream>>>(features, conf, Rws, u, w, c0, xmix);

  // 6) out = xmix Wc^T + bc   (8192 x 1024 x 1024, fp32 out -> d_out)
  gemm_nt_kernel<true><<<dim3(64, 8), 256, 0, stream>>>(xmix, Wc, out, bc,
                                                        Bdim, Hdim, Hdim);

  (void)in_sizes; (void)n_in; (void)out_size; (void)ws_size;
}
